// SE3EquivariantCrossAttention_65429531787449
// MI455X (gfx1250) — matrix operand order repaired and checked by reading, not compiled, not run on hardware
//
#include <hip/hip_runtime.h>

typedef __bf16 bf16;
typedef __attribute__((ext_vector_type(16))) __bf16 v16bf;
typedef __attribute__((ext_vector_type(8)))  __bf16 v8bf;
typedef __attribute__((ext_vector_type(8)))  float  v8f;

#define NLIG 1024
#define NPOK 8192
#define HID  256
#define ADIM 512
#define NH   4
#define HD   64
#define CUTOFF 10.0f
#define NSPLIT 8                         // j-dimension split-K factor for attention
#define NTILES 64                        // NLIG/16 ligand row tiles

union V16 { v16bf v; v8bf p8[2]; bf16 e[16]; };

#define WMMA_BF16(a, b, c) \
  __builtin_amdgcn_wmma_f32_16x16x32_bf16(false, (a), false, (b), (short)0, (c), false, false)

static __device__ __forceinline__ v8f v8f_zero() {
  v8f z = {0.f, 0.f, 0.f, 0.f, 0.f, 0.f, 0.f, 0.f};
  return z;
}

// ---------------------------------------------------------------------------
// Convert inputs/weights to bf16; weights stored transposed: wT[n][k] = w[k][n]
// ---------------------------------------------------------------------------
__global__ void k_convert(const float* __restrict__ hlig, const float* __restrict__ hatom,
                          const float* __restrict__ qw, const float* __restrict__ kw,
                          const float* __restrict__ vw, const float* __restrict__ ow,
                          const float* __restrict__ c1w,
                          bf16* __restrict__ hligb, bf16* __restrict__ hatomb,
                          bf16* __restrict__ qwT, bf16* __restrict__ kwT,
                          bf16* __restrict__ vwT, bf16* __restrict__ owT,
                          bf16* __restrict__ c1wT)
{
  const int stride = gridDim.x * blockDim.x;
  const int t0 = blockIdx.x * blockDim.x + threadIdx.x;
  for (int i = t0; i < NLIG * HID; i += stride) hligb[i] = (bf16)hlig[i];
  for (int i = t0; i < NPOK * ADIM; i += stride) hatomb[i] = (bf16)hatom[i];
  for (int i = t0; i < HID * HID; i += stride) {
    const int k = i >> 8, n = i & 255;
    qwT[n * HID + k]  = (bf16)qw[i];
    owT[n * HID + k]  = (bf16)ow[i];
    c1wT[n * HID + k] = (bf16)c1w[i];
  }
  for (int i = t0; i < ADIM * HID; i += stride) {
    const int k = i >> 8, n = i & 255;
    kwT[n * ADIM + k] = (bf16)kw[i];
    vwT[n * ADIM + k] = (bf16)vw[i];
  }
}

// ---------------------------------------------------------------------------
// Generic bf16 WMMA GEMM: out[M,N] = A[M,K] @ W[K,N] + bias  (WT[n][k] layout)
// flags: 1 = silu epilogue, 2 = store bf16 output transposed out[c*M + r]
// ---------------------------------------------------------------------------
__global__ __launch_bounds__(256) void k_gemm(
    const bf16* __restrict__ A, const bf16* __restrict__ WT,
    const float* __restrict__ bias, int M, int K, int N,
    bf16* __restrict__ out_bf, float* __restrict__ out_f,
    const float* __restrict__ residual, float scale, int flags)
{
  const int w = threadIdx.x >> 5, lane = threadIdx.x & 31;
  const int n = lane & 15, hi = lane >> 4;
  const int tilesN = N >> 4;
  const int tile = blockIdx.x * 8 + w;
  if (tile >= (M >> 4) * tilesN) return;            // wave-uniform guard
  const int row0 = (tile / tilesN) << 4;
  const int col0 = (tile % tilesN) << 4;

  // A tile 16x32: elems e -> K = hi*16+e (matches B)
  const bf16* arow = A + (size_t)(row0 + n) * K + hi * 16;
  // B tile 32x16 (contiguous k order, as A): elems e -> K = hi*16+e, N = lane&15
  const bf16* bcol = WT + (size_t)(col0 + n) * K + hi * 16;

  v8f acc = v8f_zero();
  for (int kk = 0; kk < K; kk += 32) {
    V16 a;
    a.p8[0] = *(const v8bf*)(arow + kk);
    a.p8[1] = *(const v8bf*)(arow + kk + 8);
    const v16bf b = *(const v16bf*)(bcol + kk);
    acc = WMMA_BF16(a.v, b, acc);
  }
#pragma unroll
  for (int v = 0; v < 8; ++v) {
    const int r = row0 + v + hi * 8;                // C layout: row = v + hi*8
    const int c = col0 + n;                         // col = lane&15
    float val = (acc[v] + bias[c]) * scale;
    if (flags & 1) val = val / (1.f + __expf(-val));      // silu
    if (residual) val += residual[(size_t)r * N + c];
    if (out_f) out_f[(size_t)r * N + c] = val;
    if (out_bf) {
      const size_t idx = (flags & 2) ? ((size_t)c * M + r) : ((size_t)r * N + c);
      out_bf[idx] = (bf16)val;
    }
  }
}

// ---------------------------------------------------------------------------
// Fused cross-attention, split-K over pocket atoms for occupancy:
// grid = (64 i-tiles, NSPLIT j-ranges), 8 waves/block. Each block emits
// UNNORMALIZED flash partials (m, l, O, cm) for its 1024-atom j-range.
// Waves 0..3 own head h (scores WMMA + online softmax + coord partials);
// wave (h, ch) accumulates O columns [h*64 + ch*32, +32).
// ---------------------------------------------------------------------------
__global__ __launch_bounds__(256) void k_attn(
    const bf16* __restrict__ Qbf, const bf16* __restrict__ Kbf,
    const bf16* __restrict__ VTbf,
    const float* __restrict__ xlig, const float* __restrict__ xpok,
    const float* __restrict__ e1w, const float* __restrict__ e1b,
    const float* __restrict__ e2w, const float* __restrict__ e2b,
    float* __restrict__ pO,   // [NTILES*NSPLIT][16][256]
    float* __restrict__ pm,   // [NTILES*NSPLIT][4][16]
    float* __restrict__ pl,   // [NTILES*NSPLIT][4][16]
    float* __restrict__ pcm)  // [NTILES*NSPLIT][4][16][3]
{
  __shared__ float s_e1w[256], s_e1b[256], s_e2w[1024];
  __shared__ float s_xl[16][3];
  __shared__ float s_rel[3][16][32];
  __shared__ float s_rinv[16][32];
  __shared__ float s_d[16][32];
  __shared__ float s_bias[4][16][32];
  __shared__ bf16  s_P[4][16][32];
  __shared__ float s_alpha[4][16];
  __shared__ int   s_any;

  const int tid  = threadIdx.x;
  const int w    = tid >> 5;
  const int lane = tid & 31;
  const int hh   = w & 3;        // head
  const int ch   = w >> 2;       // V column half
  const int n    = lane & 15;
  const int hi   = lane >> 4;
  const int it   = blockIdx.x;
  const int sp   = blockIdx.y;
  const int i0   = it << 4;
  const size_t part = (size_t)it * NSPLIT + sp;

  for (int i = tid; i < 256; i += 256) { s_e1w[i] = e1w[i]; s_e1b[i] = e1b[i]; }
  for (int i = tid; i < 1024; i += 256) s_e2w[i] = e2w[i];
  if (tid < 48) s_xl[tid / 3][tid % 3] = xlig[(i0 + tid / 3) * 3 + tid % 3];
  const float eb0 = e2b[0], eb1 = e2b[1], eb2 = e2b[2], eb3 = e2b[3];

  // Leader waves preload Q tile (16 x 64 per head) in WMMA A layout
  V16 aq0, aq1;
  if (ch == 0) {
    const bf16* qp = Qbf + (size_t)(i0 + n) * HID + hh * HD + hi * 16;
    aq0.p8[0] = *(const v8bf*)(qp);
    aq0.p8[1] = *(const v8bf*)(qp + 8);
    aq1.p8[0] = *(const v8bf*)(qp + 32);
    aq1.p8[1] = *(const v8bf*)(qp + 40);
  }

  float m_run[8], l_run[8], cmx[8], cmy[8], cmz[8];
  v8f o0 = v8f_zero(), o1 = v8f_zero();
#pragma unroll
  for (int v = 0; v < 8; ++v) {
    m_run[v] = -__builtin_inff(); l_run[v] = 0.f;
    cmx[v] = cmy[v] = cmz[v] = 0.f;
  }

  const int pi = tid & 15;   // pair row (ligand-local)
  const int pj = tid >> 4;   // pair col base (0..15); second pair at +16

  __syncthreads();

  const int jt0 = sp * (NPOK / 32 / NSPLIT);
  const int jt1 = jt0 + (NPOK / 32 / NSPLIT);
  for (int jt = jt0; jt < jt1; ++jt) {
    const int j0 = jt << 5;
    __syncthreads();                       // protects s_P/s_alpha from prev iter
    if (tid == 0) s_any = 0;
    __syncthreads();

    // ---- distances for 16x32 pair tile (2 pairs per thread) ----
    int any = 0;
#pragma unroll
    for (int pp = 0; pp < 2; ++pp) {
      const int jj = pj + pp * 16;
      const float dx = xpok[(j0 + jj) * 3 + 0] - s_xl[pi][0];
      const float dy = xpok[(j0 + jj) * 3 + 1] - s_xl[pi][1];
      const float dz = xpok[(j0 + jj) * 3 + 2] - s_xl[pi][2];
      const float d  = sqrtf(dx * dx + dy * dy + dz * dz);
      s_rel[0][pi][jj] = dx; s_rel[1][pi][jj] = dy; s_rel[2][pi][jj] = dz;
      s_rinv[pi][jj] = 1.f / (d + 1e-8f);
      s_d[pi][jj] = d;
      any |= (d < CUTOFF) ? 1 : 0;
    }
    if (__any(any) && lane == 0) s_any = 1;
    __syncthreads();
    if (!s_any) continue;                  // block-uniform tile skip (~85% of tiles)

    // ---- edge-bias MLP (shared across heads), only for unmasked pairs ----
#pragma unroll
    for (int pp = 0; pp < 2; ++pp) {
      const int jj = pj + pp * 16;
      const float d = s_d[pi][jj];
      float a0 = 0.f, a1 = 0.f, a2 = 0.f, a3 = 0.f;
      if (d < CUTOFF) {
#pragma unroll 8
        for (int c = 0; c < 256; ++c) {
          const float t = fmaf(d, s_e1w[c], s_e1b[c]);
          const float s = t / (1.f + __expf(-t));           // silu
          a0 = fmaf(s, s_e2w[c * 4 + 0], a0);
          a1 = fmaf(s, s_e2w[c * 4 + 1], a1);
          a2 = fmaf(s, s_e2w[c * 4 + 2], a2);
          a3 = fmaf(s, s_e2w[c * 4 + 3], a3);
        }
      }
      s_bias[0][pi][jj] = a0 + eb0;
      s_bias[1][pi][jj] = a1 + eb1;
      s_bias[2][pi][jj] = a2 + eb2;
      s_bias[3][pi][jj] = a3 + eb3;
    }
    __syncthreads();

    // ---- leader waves: scores (WMMA) + online softmax for head hh ----
    if (ch == 0) {
      v8f sc0 = v8f_zero(), sc1 = v8f_zero();
      const bf16* kp0 = Kbf + (size_t)(j0 + n) * HID + hh * HD + hi * 16;
      const bf16* kp1 = kp0 + (size_t)16 * HID;
      sc0 = WMMA_BF16(aq0.v, *(const v16bf*)(kp0),      sc0);
      sc0 = WMMA_BF16(aq1.v, *(const v16bf*)(kp0 + 32), sc0);
      sc1 = WMMA_BF16(aq0.v, *(const v16bf*)(kp1),      sc1);
      sc1 = WMMA_BF16(aq1.v, *(const v16bf*)(kp1 + 32), sc1);

      float sv0[8], sv1[8], rm[8];
#pragma unroll
      for (int v = 0; v < 8; ++v) {
        const int r = v + hi * 8;
        float x0 = sc0[v] + s_bias[hh][r][n];
        float x1 = sc1[v] + s_bias[hh][r][16 + n];
        if (s_d[r][n]      >= CUTOFF) x0 = -__builtin_inff();
        if (s_d[r][16 + n] >= CUTOFF) x1 = -__builtin_inff();
        sv0[v] = x0; sv1[v] = x1;
        rm[v] = fmaxf(x0, x1);
      }
#pragma unroll
      for (int off = 1; off < 16; off <<= 1)
#pragma unroll
        for (int v = 0; v < 8; ++v)
          rm[v] = fmaxf(rm[v], __shfl_xor(rm[v], off, 32));

      float rs[8], al[8];
#pragma unroll
      for (int v = 0; v < 8; ++v) {
        const int r = v + hi * 8;
        const float mnew = fmaxf(m_run[v], rm[v]);
        const bool dead = (mnew == -__builtin_inff());
        const float alpha = dead ? 1.f : __expf(m_run[v] - mnew);
        const float p0 = dead ? 0.f : __expf(sv0[v] - mnew);
        const float p1 = dead ? 0.f : __expf(sv1[v] - mnew);
        m_run[v] = mnew; al[v] = alpha; rs[v] = p0 + p1;
        const float u0 = p0 * s_rinv[r][n];
        const float u1 = p1 * s_rinv[r][16 + n];
        cmx[v] = fmaf(cmx[v], alpha, u0 * s_rel[0][r][n] + u1 * s_rel[0][r][16 + n]);
        cmy[v] = fmaf(cmy[v], alpha, u0 * s_rel[1][r][n] + u1 * s_rel[1][r][16 + n]);
        cmz[v] = fmaf(cmz[v], alpha, u0 * s_rel[2][r][n] + u1 * s_rel[2][r][16 + n]);
        s_P[hh][r][n]      = (bf16)p0;
        s_P[hh][r][16 + n] = (bf16)p1;
      }
#pragma unroll
      for (int off = 1; off < 16; off <<= 1)
#pragma unroll
        for (int v = 0; v < 8; ++v)
          rs[v] += __shfl_xor(rs[v], off, 32);
#pragma unroll
      for (int v = 0; v < 8; ++v) l_run[v] = fmaf(l_run[v], al[v], rs[v]);
      if (n == 0)
#pragma unroll
        for (int v = 0; v < 8; ++v) s_alpha[hh][v + hi * 8] = al[v];
    }
    __syncthreads();

    // ---- all waves: O = O*alpha + P @ V  (WMMA, K = 32 pocket atoms) ----
    V16 pa;                                // P tile in A layout from LDS
#pragma unroll
    for (int e = 0; e < 8; ++e) pa.e[e]     = s_P[hh][n][hi * 16 + e];
#pragma unroll
    for (int e = 0; e < 8; ++e) pa.e[8 + e] = s_P[hh][n][hi * 16 + 8 + e];
    const int c0 = hh * HD + ch * 32;
    const bf16* vp = VTbf + (size_t)(c0 + n) * NPOK + j0 + hi * 16;
    const v16bf bv0 = *(const v16bf*)(vp);
    const v16bf bv1 = *(const v16bf*)(vp + (size_t)16 * NPOK);
#pragma unroll
    for (int v = 0; v < 8; ++v) {
      const float a = s_alpha[hh][v + hi * 8];
      o0[v] *= a; o1[v] *= a;
    }
    o0 = WMMA_BF16(pa.v, bv0, o0);
    o1 = WMMA_BF16(pa.v, bv1, o1);
  }

  // ---- emit unnormalized partials for the combine kernel ----
  if (ch == 0) {
#pragma unroll
    for (int off = 1; off < 16; off <<= 1)
#pragma unroll
      for (int v = 0; v < 8; ++v) {
        cmx[v] += __shfl_xor(cmx[v], off, 32);
        cmy[v] += __shfl_xor(cmy[v], off, 32);
        cmz[v] += __shfl_xor(cmz[v], off, 32);
      }
    if (n == 0)
#pragma unroll
      for (int v = 0; v < 8; ++v) {
        const int r = v + hi * 8;
        const size_t b = (part * 4 + hh) * 16 + r;
        pm[b] = m_run[v];
        pl[b] = l_run[v];
        pcm[b * 3 + 0] = cmx[v];
        pcm[b * 3 + 1] = cmy[v];
        pcm[b * 3 + 2] = cmz[v];
      }
  }
  {
    const int c0 = hh * HD + ch * 32;
#pragma unroll
    for (int v = 0; v < 8; ++v) {
      const int r = v + hi * 8;
      float* po = pO + (part * 16 + r) * HID + c0;
      po[n]      = o0[v];
      po[16 + n] = o1[v];
    }
  }
}

// ---------------------------------------------------------------------------
// Merge split-K flash partials: standard exp(m_s - M) reweighting.
// One block per i-tile.
// ---------------------------------------------------------------------------
__global__ __launch_bounds__(256) void k_combine(
    const float* __restrict__ pO, const float* __restrict__ pm,
    const float* __restrict__ pl, const float* __restrict__ pcm,
    bf16* __restrict__ hatt, float* __restrict__ coord_msg)
{
  __shared__ float s_L[4][16];
  __shared__ float s_w[NSPLIT][4][16];
  const int tid = threadIdx.x;
  const int it = blockIdx.x;
  const int i0 = it << 4;

  if (tid < 64) {
    const int h = tid >> 4, r = tid & 15;
    float M = -__builtin_inff();
#pragma unroll
    for (int s = 0; s < NSPLIT; ++s)
      M = fmaxf(M, pm[((size_t)(it * NSPLIT + s) * 4 + h) * 16 + r]);
    float L = 0.f;
#pragma unroll
    for (int s = 0; s < NSPLIT; ++s) {
      const size_t b = ((size_t)(it * NSPLIT + s) * 4 + h) * 16 + r;
      const float wgt = (M == -__builtin_inff()) ? 0.f : __expf(pm[b] - M);
      s_w[s][h][r] = wgt;
      L += wgt * pl[b];
    }
    s_L[h][r] = L;
  }
  __syncthreads();

  for (int idx = tid; idx < 16 * HID; idx += 256) {
    const int r = idx >> 8, c = idx & 255, h = c >> 6;
    float sum = 0.f;
#pragma unroll
    for (int s = 0; s < NSPLIT; ++s)
      sum += s_w[s][h][r] * pO[((size_t)(it * NSPLIT + s) * 16 + r) * HID + c];
    const float L = s_L[h][r];
    hatt[(size_t)(i0 + r) * HID + c] = (bf16)((L > 0.f) ? sum / L : 0.f);
  }

  if (tid < 48) {
    const int r = tid / 3, d = tid % 3;
    float cm = 0.f;
#pragma unroll
    for (int h = 0; h < 4; ++h) {
      const float L = s_L[h][r];
      if (L > 0.f) {
        float acc = 0.f;
#pragma unroll
        for (int s = 0; s < NSPLIT; ++s)
          acc += s_w[s][h][r] * pcm[(((size_t)(it * NSPLIT + s) * 4 + h) * 16 + r) * 3 + d];
        cm += acc / L;
      }
    }
    coord_msg[(i0 + r) * 3 + d] = 0.25f * cm;
  }
}

// ---------------------------------------------------------------------------
// coord_scale = t1 @ c2w + c2b ; x_out = x_lig + coord_scale * coord_msg
// One wave per ligand row.
// ---------------------------------------------------------------------------
__global__ __launch_bounds__(256) void k_coordfinal(
    const bf16* __restrict__ t1b, const float* __restrict__ c2w,
    const float* __restrict__ c2b, const float* __restrict__ xlig,
    const float* __restrict__ cmsg, float* __restrict__ xout)
{
  const int w = threadIdx.x >> 5, lane = threadIdx.x & 31;
  const int row = blockIdx.x * 8 + w;
  if (row >= NLIG) return;
  float s = 0.f;
  for (int k = lane; k < HID; k += 32)
    s += (float)t1b[(size_t)row * HID + k] * c2w[k];
#pragma unroll
  for (int off = 16; off >= 1; off >>= 1) s += __shfl_xor(s, off, 32);
  s += c2b[0];
  if (lane < 3)
    xout[row * 3 + lane] = xlig[row * 3 + lane] + s * cmsg[row * 3 + lane];
}

// ---------------------------------------------------------------------------
extern "C" void kernel_launch(void* const* d_in, const int* in_sizes, int n_in,
                              void* d_out, int out_size, void* d_ws, size_t ws_size,
                              hipStream_t stream)
{
  const float* h_lig  = (const float*)d_in[0];
  const float* x_lig  = (const float*)d_in[1];
  const float* h_atom = (const float*)d_in[2];
  const float* x_pok  = (const float*)d_in[3];
  const float* qw  = (const float*)d_in[4];  const float* qb  = (const float*)d_in[5];
  const float* kw  = (const float*)d_in[6];  const float* kb  = (const float*)d_in[7];
  const float* vw  = (const float*)d_in[8];  const float* vb  = (const float*)d_in[9];
  const float* ow  = (const float*)d_in[10]; const float* ob  = (const float*)d_in[11];
  const float* e1w = (const float*)d_in[12]; const float* e1b = (const float*)d_in[13];
  const float* e2w = (const float*)d_in[14]; const float* e2b = (const float*)d_in[15];
  const float* c1w = (const float*)d_in[16]; const float* c1b = (const float*)d_in[17];
  const float* c2w = (const float*)d_in[18]; const float* c2b = (const float*)d_in[19];

  char* ws = (char*)d_ws;
  size_t off = 0;
  auto take = [&](size_t bytes) -> char* {
    char* p = ws + off;
    off = (off + bytes + 255) & ~(size_t)255;
    return p;
  };
  bf16* hligb  = (bf16*)take((size_t)NLIG * HID * 2);
  bf16* hatomb = (bf16*)take((size_t)NPOK * ADIM * 2);
  bf16* qwT    = (bf16*)take((size_t)HID * HID * 2);
  bf16* kwT    = (bf16*)take((size_t)ADIM * HID * 2);
  bf16* vwT    = (bf16*)take((size_t)ADIM * HID * 2);
  bf16* owT    = (bf16*)take((size_t)HID * HID * 2);
  bf16* c1wT   = (bf16*)take((size_t)HID * HID * 2);
  bf16* Qbf    = (bf16*)take((size_t)NLIG * HID * 2);
  bf16* Kbf    = (bf16*)take((size_t)NPOK * HID * 2);
  bf16* VTbf   = (bf16*)take((size_t)HID * NPOK * 2);
  bf16* hattb  = (bf16*)take((size_t)NLIG * HID * 2);
  bf16* t1b    = (bf16*)take((size_t)NLIG * HID * 2);
  float* cmsg  = (float*)take((size_t)NLIG * 3 * 4);
  float* pO    = (float*)take((size_t)NTILES * NSPLIT * 16 * HID * 4);   // 8 MB
  float* pm    = (float*)take((size_t)NTILES * NSPLIT * 4 * 16 * 4);
  float* pl    = (float*)take((size_t)NTILES * NSPLIT * 4 * 16 * 4);
  float* pcm   = (float*)take((size_t)NTILES * NSPLIT * 4 * 16 * 3 * 4);

  float* h_out = (float*)d_out;
  float* x_out = h_out + (size_t)NLIG * HID;

  k_convert<<<512, 256, 0, stream>>>(h_lig, h_atom, qw, kw, vw, ow, c1w,
                                     hligb, hatomb, qwT, kwT, vwT, owT, c1wT);

  // Projections (1/sqrt(HEAD_DIM)=0.125 folded into Q); V stored transposed.
  k_gemm<<<(NLIG / 16) * (HID / 16) / 8, 256, 0, stream>>>(
      hligb, qwT, qb, NLIG, HID, HID, Qbf, nullptr, nullptr, 0.125f, 0);
  k_gemm<<<(NPOK / 16) * (HID / 16) / 8, 256, 0, stream>>>(
      hatomb, kwT, kb, NPOK, ADIM, HID, Kbf, nullptr, nullptr, 1.f, 0);
  k_gemm<<<(NPOK / 16) * (HID / 16) / 8, 256, 0, stream>>>(
      hatomb, vwT, vb, NPOK, ADIM, HID, VTbf, nullptr, nullptr, 1.f, 2);

  // Split-K fused attention: 64 x 8 = 512 blocks (4096 waves) to fill the WGPs.
  k_attn<<<dim3(NTILES, NSPLIT), 256, 0, stream>>>(
      Qbf, Kbf, VTbf, x_lig, x_pok, e1w, e1b, e2w, e2b, pO, pm, pl, pcm);
  k_combine<<<NTILES, 256, 0, stream>>>(pO, pm, pl, pcm, hattb, cmsg);

  // h_out = h_lig + h_att @ ow + ob  (f32 output)
  k_gemm<<<(NLIG / 16) * (HID / 16) / 8, 256, 0, stream>>>(
      hattb, owT, ob, NLIG, HID, HID, nullptr, h_out, h_lig, 1.f, 0);
  // t1 = silu(h_att @ c1w + c1b)  (bf16)
  k_gemm<<<(NLIG / 16) * (HID / 16) / 8, 256, 0, stream>>>(
      hattb, c1wT, c1b, NLIG, HID, HID, t1b, nullptr, nullptr, 1.f, 1);

  k_coordfinal<<<NLIG / 8, 256, 0, stream>>>(t1b, c2w, c2b, x_lig, cmsg, x_out);
}
